// MoELinear_55473797595878
// MI455X (gfx1250) — compile-verified, matrix operand order checked by reading, not run on hardware
//
#include <hip/hip_runtime.h>
#include <hip/hip_bf16.h>
#include <math.h>

#define T_TOK 8192
#define IN_D 2048
#define OUT_D 2048
#define PROJ_D 512
#define NEXP 8
#define TOPK 2
#define M_BLK 64            // rows per block (4 x 16 subtiles per wave)

// LDS B-tile row pitch: 32 data elems + 4 pad elems (TDM pad: 2 dwords per 16-dword row)
#define LDSROW 36
#define BUF_ELEMS (128 * LDSROW)

#if __has_builtin(__builtin_amdgcn_tensor_load_to_lds)
#define HAVE_TDM 1
#else
#define HAVE_TDM 0
#endif

typedef __attribute__((ext_vector_type(16))) __bf16 v16bf;
typedef __attribute__((ext_vector_type(8)))  float  v8f;
typedef __attribute__((ext_vector_type(4)))  unsigned int v4u;
typedef __attribute__((ext_vector_type(8)))  int v8i;
typedef __attribute__((ext_vector_type(4)))  int v4i;

__device__ __forceinline__ unsigned short f2bf(float f) {
  unsigned u = __builtin_bit_cast(unsigned, f);
  u += 0x7FFFu + ((u >> 16) & 1u);   // round-to-nearest-even
  return (unsigned short)(u >> 16);
}
__device__ __forceinline__ float bf2f(unsigned short h) {
  unsigned u = ((unsigned)h) << 16;
  return __builtin_bit_cast(float, u);
}
__device__ __forceinline__ float gelu_tanh(float v) {
  return 0.5f * v * (1.0f + tanhf(0.7978845608028654f * (v + 0.044715f * v * v * v)));
}

// ---- TDM: issue async load of a 2D tile (32 x 128, 2-byte elems) into LDS
// with row padding (72B pitch). No wait — caller synchronizes with tensorcnt.
__device__ __forceinline__ void tdm_issue_tile(const unsigned short* gsrc,
                                               unsigned lds_byte_addr,
                                               unsigned tensorX, unsigned tensorY,
                                               unsigned strideElems) {
#if HAVE_TDM
  unsigned long long ga = (unsigned long long)gsrc;
  // D# group 0: count=1 | lds_addr | global_addr[56:0] | type=2 ("image")
  v4u g0 = { 1u,
             lds_byte_addr,
             (unsigned)ga,
             (unsigned)((ga >> 32) & 0x01FFFFFFu) | (2u << 30) };
  // D# group 1: data_size=2B | pad_enable | pad_interval=16dw | pad_amount=2dw,
  // tensor dims, tile 32x128, dim0 stride
  v8i g1 = { (int)((1u << 16) | (1u << 20) | (3u << 22) | (1u << 25)),
             (int)((tensorX & 0xFFFFu) << 16),
             (int)((tensorX >> 16) | ((tensorY & 0xFFFFu) << 16)),
             (int)((tensorY >> 16) | (32u << 16)),
             (int)128u,
             (int)strideElems,
             0, 0 };
  v4i g2 = {0, 0, 0, 0};
  v4i g3 = {0, 0, 0, 0};
  v8i g4 = {0, 0, 0, 0, 0, 0, 0, 0};
  __builtin_amdgcn_tensor_load_to_lds(g0, g1, g2, g3, g4, 0);
#else
  (void)gsrc; (void)lds_byte_addr; (void)tensorX; (void)tensorY; (void)strideElems;
#endif
}

// ---------------- init: zero expert counters ----------------
__global__ void k_init(int* __restrict__ cnt) {
  if (threadIdx.x < NEXP) cnt[threadIdx.x] = 0;
}

// ---------------- f32 -> bf16 converts (w1/w2 transposed to K-contiguous) ----
__global__ void k_conv_x(const float* __restrict__ x, unsigned short* __restrict__ xb) {
  const int n = T_TOK * IN_D;
  for (int i = blockIdx.x * blockDim.x + threadIdx.x; i < n; i += gridDim.x * blockDim.x)
    xb[i] = f2bf(x[i]);
}
__global__ void k_conv_w1(const float* __restrict__ w1, unsigned short* __restrict__ w1t) {
  const int n = NEXP * IN_D * PROJ_D;
  for (int idx = blockIdx.x * blockDim.x + threadIdx.x; idx < n; idx += gridDim.x * blockDim.x) {
    int e = idx / (IN_D * PROJ_D);
    int r = idx - e * (IN_D * PROJ_D);
    int i = r / PROJ_D;
    int p = r - i * PROJ_D;
    w1t[((size_t)e * PROJ_D + p) * IN_D + i] = f2bf(w1[idx]);  // [e][p][i]
  }
}
__global__ void k_conv_w2(const float* __restrict__ w2, unsigned short* __restrict__ w2t) {
  const int n = NEXP * PROJ_D * OUT_D;
  for (int idx = blockIdx.x * blockDim.x + threadIdx.x; idx < n; idx += gridDim.x * blockDim.x) {
    int e = idx / (PROJ_D * OUT_D);
    int r = idx - e * (PROJ_D * OUT_D);
    int p = r / OUT_D;
    int o = r - p * OUT_D;
    w2t[((size_t)e * OUT_D + o) * PROJ_D + p] = f2bf(w2[idx]); // [e][o][p]
  }
}

// ---------------- gating: one wave32 per token ----------------
__global__ __launch_bounds__(256) void k_gate(
    const float* __restrict__ x, const float* __restrict__ gw, const float* __restrict__ gb,
    int* __restrict__ cnt, int* __restrict__ expert_of, int* __restrict__ pos_of,
    float* __restrict__ wgt, int* __restrict__ token_of) {
  int lane = threadIdx.x & 31;
  int wave = threadIdx.x >> 5;
  int t = blockIdx.x * 8 + wave;
  if (t >= T_TOK) return;
  const float* xr = x + (size_t)t * IN_D;
  float acc[NEXP];
#pragma unroll
  for (int e = 0; e < NEXP; ++e) acc[e] = 0.f;
  for (int i = lane; i < IN_D; i += 32) {
    float xv = xr[i];
    const float* g = gw + (size_t)i * NEXP;
#pragma unroll
    for (int e = 0; e < NEXP; ++e) acc[e] += xv * g[e];
  }
#pragma unroll
  for (int e = 0; e < NEXP; ++e) {
#pragma unroll
    for (int off = 16; off > 0; off >>= 1) acc[e] += __shfl_xor(acc[e], off, 32);
  }
  if (lane == 0) {
    const float scale = 0.022097086912079608f;  // 1/sqrt(2048)
    float l[NEXP], lmax = -1e30f;
#pragma unroll
    for (int e = 0; e < NEXP; ++e) { l[e] = (acc[e] + gb[e]) * scale; lmax = fmaxf(lmax, l[e]); }
    float sum = 0.f;
#pragma unroll
    for (int e = 0; e < NEXP; ++e) { l[e] = expf(l[e] - lmax); sum += l[e]; }
    float inv = 1.f / sum;
    float p[NEXP];
#pragma unroll
    for (int e = 0; e < NEXP; ++e) p[e] = l[e] * inv;
    int i0 = 0;
#pragma unroll
    for (int e = 1; e < NEXP; ++e) if (p[e] > p[i0]) i0 = e;
    int i1 = (i0 == 0) ? 1 : 0;
#pragma unroll
    for (int e = 0; e < NEXP; ++e) if (e != i0 && p[e] > p[i1]) i1 = e;
    int sel[TOPK] = {i0, i1};
#pragma unroll
    for (int k = 0; k < TOPK; ++k) {
      int e = sel[k];
      int pos = atomicAdd(&cnt[e], 1);
      expert_of[t * TOPK + k] = e;
      pos_of[t * TOPK + k] = pos;
      wgt[t * TOPK + k] = p[e];
      token_of[e * T_TOK + pos] = t;
    }
  }
}

// ---------------- tiny prefix sum over E=8 counts ----------------
__global__ void k_prefix(const int* __restrict__ cnt, int* __restrict__ base) {
  if (threadIdx.x == 0 && blockIdx.x == 0) {
    int b = 0;
    for (int e = 0; e < NEXP; ++e) { base[e] = b; b += cnt[e]; }
  }
}

// Cooperative global->LDS fallback copy (only used if TDM builtin is absent)
__device__ __forceinline__ void lds_fill_fallback(const unsigned short* gsrc,
                                                  unsigned short* Bsh,
                                                  unsigned strideElems) {
#if !HAVE_TDM
  for (int i = threadIdx.x; i < 128 * 8; i += 256) {  // 8 b64 chunks per row
    int row = i >> 3;
    int c4 = (i & 7) * 4;
    *(uint2*)&Bsh[row * LDSROW + c4] = *(const uint2*)&gsrc[(size_t)row * strideElems + c4];
  }
#else
  (void)gsrc; (void)Bsh; (void)strideElems;
#endif
}

// ---------------- GEMM1: h = gelu(X_e @ w1[e] + b1[e]) ----------------
// grid = (PROJ/128, T/M_BLK, E), block = 256 (8 waves; each wave: 16 cols x 64 rows)
__global__ __launch_bounds__(256) void k_gemm1(
    const unsigned short* __restrict__ xb, const unsigned short* __restrict__ w1t,
    const float* __restrict__ b1, const int* __restrict__ token_of,
    const int* __restrict__ cnt, const int* __restrict__ base,
    unsigned short* __restrict__ hb) {
  __shared__ unsigned short Bsh[2][BUF_ELEMS];
  int e = blockIdx.z;
  int cnt_e = cnt[e];
  int rowBase = blockIdx.y * M_BLK;
  if (rowBase >= cnt_e) return;
  int lane = threadIdx.x & 31;
  int wave = threadIdx.x >> 5;
  int r16 = lane & 15;
  int hi = lane >> 4;         // 0: K 0..7 half, 1: K 8..15 half
  int kb = hi * 8;
  const unsigned short* aptr[4];
#pragma unroll
  for (int m = 0; m < 4; ++m) {
    int row = rowBase + m * 16 + r16;
    int tok = token_of[e * T_TOK + (row < cnt_e ? row : 0)];
    aptr[m] = xb + (size_t)tok * IN_D;
  }
  int colBase = blockIdx.x * 128;
  int col = colBase + wave * 16 + r16;
  int col_local = wave * 16 + r16;
  const unsigned short* btile0 = w1t + ((size_t)e * PROJ_D + colBase) * IN_D;
  unsigned lds0 = (unsigned)(unsigned long long)&Bsh[0][0];
  v8f acc[4];
#pragma unroll
  for (int m = 0; m < 4; ++m) acc[m] = {};
  const int NCHUNK = IN_D / 32;
  if (wave == 0) tdm_issue_tile(btile0, lds0, IN_D, PROJ_D, IN_D);
  for (int chunk = 0; chunk < NCHUNK; ++chunk) {
    int cur = chunk & 1;
#if HAVE_TDM
    if (wave == 0) __builtin_amdgcn_s_wait_tensorcnt(0);
    __syncthreads();   // publish buf[cur]; proves buf[cur^1] reads of prev chunk done
    if (wave == 0 && chunk + 1 < NCHUNK)
      tdm_issue_tile(btile0 + (chunk + 1) * 32, lds0 + (cur ^ 1) * (BUF_ELEMS * 2),
                     IN_D, PROJ_D, IN_D);
#else
    __syncthreads();
    lds_fill_fallback(btile0 + chunk * 32, &Bsh[cur][0], IN_D);
    __syncthreads();
#endif
    union { v16bf v; unsigned u[8]; } b;
#pragma unroll
    for (int i = 0; i < 8; ++i) {
      int k0 = ((i & 4) << 2) + kb + 2 * (i & 3);
      b.u[i] = *(const unsigned*)(&Bsh[cur][col_local * LDSROW + k0]);
    }
#pragma unroll
    for (int m = 0; m < 4; ++m) {
      union { v16bf v; unsigned u[8]; } a;
#pragma unroll
      for (int i = 0; i < 8; ++i) {
        int k0 = ((i & 4) << 2) + kb + 2 * (i & 3);
        a.u[i] = *(const unsigned*)(aptr[m] + chunk * 32 + k0);
      }
      acc[m] = __builtin_amdgcn_wmma_f32_16x16x32_bf16(false, a.v, false, b.v,
                                                       (short)0, acc[m], false, false);
    }
  }
  float bias = b1[e * PROJ_D + col];
  int sbase = base[e];
#pragma unroll
  for (int m = 0; m < 4; ++m) {
#pragma unroll
    for (int r = 0; r < 8; ++r) {
      int gr = rowBase + m * 16 + r + hi * 8;
      if (gr < cnt_e) {
        float g = gelu_tanh(acc[m][r] + bias);
        hb[(size_t)(sbase + gr) * PROJ_D + col] = f2bf(g);
      }
    }
  }
}

// ---------------- GEMM2: y_sel = h @ w2[e] + b2[e] ----------------
// grid = (OUT/128, T/M_BLK, E), block = 256
__global__ __launch_bounds__(256) void k_gemm2(
    const unsigned short* __restrict__ hb, const unsigned short* __restrict__ w2t,
    const float* __restrict__ b2, const int* __restrict__ cnt, const int* __restrict__ base,
    unsigned short* __restrict__ ysel) {
  __shared__ unsigned short Bsh[2][BUF_ELEMS];
  int e = blockIdx.z;
  int cnt_e = cnt[e];
  int rowBase = blockIdx.y * M_BLK;
  if (rowBase >= cnt_e) return;
  int lane = threadIdx.x & 31;
  int wave = threadIdx.x >> 5;
  int r16 = lane & 15;
  int hi = lane >> 4;
  int kb = hi * 8;
  int sbase = base[e];
  const unsigned short* aptr[4];
#pragma unroll
  for (int m = 0; m < 4; ++m) {
    int row = rowBase + m * 16 + r16;
    if (row >= cnt_e) row = cnt_e - 1;             // safe pad read
    aptr[m] = hb + (size_t)(sbase + row) * PROJ_D;
  }
  int colBase = blockIdx.x * 128;
  int col = colBase + wave * 16 + r16;
  int col_local = wave * 16 + r16;
  const unsigned short* btile0 = w2t + ((size_t)e * OUT_D + colBase) * PROJ_D;
  unsigned lds0 = (unsigned)(unsigned long long)&Bsh[0][0];
  v8f acc[4];
#pragma unroll
  for (int m = 0; m < 4; ++m) acc[m] = {};
  const int NCHUNK = PROJ_D / 32;
  if (wave == 0) tdm_issue_tile(btile0, lds0, PROJ_D, OUT_D, PROJ_D);
  for (int chunk = 0; chunk < NCHUNK; ++chunk) {
    int cur = chunk & 1;
#if HAVE_TDM
    if (wave == 0) __builtin_amdgcn_s_wait_tensorcnt(0);
    __syncthreads();
    if (wave == 0 && chunk + 1 < NCHUNK)
      tdm_issue_tile(btile0 + (chunk + 1) * 32, lds0 + (cur ^ 1) * (BUF_ELEMS * 2),
                     PROJ_D, OUT_D, PROJ_D);
#else
    __syncthreads();
    lds_fill_fallback(btile0 + chunk * 32, &Bsh[cur][0], PROJ_D);
    __syncthreads();
#endif
    union { v16bf v; unsigned u[8]; } b;
#pragma unroll
    for (int i = 0; i < 8; ++i) {
      int k0 = ((i & 4) << 2) + kb + 2 * (i & 3);
      b.u[i] = *(const unsigned*)(&Bsh[cur][col_local * LDSROW + k0]);
    }
#pragma unroll
    for (int m = 0; m < 4; ++m) {
      union { v16bf v; unsigned u[8]; } a;
#pragma unroll
      for (int i = 0; i < 8; ++i) {
        int k0 = ((i & 4) << 2) + kb + 2 * (i & 3);
        a.u[i] = *(const unsigned*)(aptr[m] + chunk * 32 + k0);
      }
      acc[m] = __builtin_amdgcn_wmma_f32_16x16x32_bf16(false, a.v, false, b.v,
                                                       (short)0, acc[m], false, false);
    }
  }
  float bias = b2[e * OUT_D + col];
#pragma unroll
  for (int m = 0; m < 4; ++m) {
#pragma unroll
    for (int r = 0; r < 8; ++r) {
      int gr = rowBase + m * 16 + r + hi * 8;
      if (gr < cnt_e)
        ysel[(size_t)(sbase + gr) * OUT_D + col] = f2bf(acc[m][r] + bias);
    }
  }
}

// ---------------- combine: y[t] = w0*y_sel[slot0] + w1*y_sel[slot1] --------
__global__ __launch_bounds__(256) void k_combine(
    const unsigned short* __restrict__ ysel, const int* __restrict__ expert_of,
    const int* __restrict__ pos_of, const float* __restrict__ wgt,
    const int* __restrict__ base, float* __restrict__ out) {
  int t = blockIdx.x;
  int e0 = expert_of[t * TOPK + 0], e1 = expert_of[t * TOPK + 1];
  size_t s0 = (size_t)(base[e0] + pos_of[t * TOPK + 0]) * OUT_D;
  size_t s1 = (size_t)(base[e1] + pos_of[t * TOPK + 1]) * OUT_D;
  float w0 = wgt[t * TOPK + 0], w1 = wgt[t * TOPK + 1];
  for (int c = threadIdx.x; c < OUT_D; c += blockDim.x)
    out[(size_t)t * OUT_D + c] = w0 * bf2f(ysel[s0 + c]) + w1 * bf2f(ysel[s1 + c]);
}

// ---------------- workspace layout ----------------
constexpr size_t OFF_XB    = 0;
constexpr size_t OFF_W1T   = OFF_XB   + (size_t)T_TOK * IN_D * 2;
constexpr size_t OFF_W2T   = OFF_W1T  + (size_t)NEXP * IN_D * PROJ_D * 2;
constexpr size_t OFF_HB    = OFF_W2T  + (size_t)NEXP * PROJ_D * OUT_D * 2;
constexpr size_t OFF_YSEL  = OFF_HB   + (size_t)T_TOK * TOPK * PROJ_D * 2;
constexpr size_t OFF_EXP   = OFF_YSEL + (size_t)T_TOK * TOPK * OUT_D * 2;
constexpr size_t OFF_POS   = OFF_EXP  + (size_t)T_TOK * TOPK * 4;
constexpr size_t OFF_WGT   = OFF_POS  + (size_t)T_TOK * TOPK * 4;
constexpr size_t OFF_TOKOF = OFF_WGT  + (size_t)T_TOK * TOPK * 4;
constexpr size_t OFF_CNT   = OFF_TOKOF + (size_t)NEXP * T_TOK * 4;
constexpr size_t OFF_BASE  = OFF_CNT  + 256;

extern "C" void kernel_launch(void* const* d_in, const int* in_sizes, int n_in,
                              void* d_out, int out_size, void* d_ws, size_t ws_size,
                              hipStream_t stream) {
  const float* x      = (const float*)d_in[0];
  const float* gate_w = (const float*)d_in[1];
  const float* gate_b = (const float*)d_in[2];
  const float* w1     = (const float*)d_in[3];
  const float* b1     = (const float*)d_in[4];
  const float* w2     = (const float*)d_in[5];
  const float* b2     = (const float*)d_in[6];
  float* out = (float*)d_out;

  char* ws = (char*)d_ws;
  unsigned short* xb    = (unsigned short*)(ws + OFF_XB);
  unsigned short* w1t   = (unsigned short*)(ws + OFF_W1T);
  unsigned short* w2t   = (unsigned short*)(ws + OFF_W2T);
  unsigned short* hb    = (unsigned short*)(ws + OFF_HB);
  unsigned short* ysel  = (unsigned short*)(ws + OFF_YSEL);
  int*   expert_of = (int*)(ws + OFF_EXP);
  int*   pos_of    = (int*)(ws + OFF_POS);
  float* wgt       = (float*)(ws + OFF_WGT);
  int*   token_of  = (int*)(ws + OFF_TOKOF);
  int*   cnt       = (int*)(ws + OFF_CNT);
  int*   base      = (int*)(ws + OFF_BASE);

  k_init<<<1, 32, 0, stream>>>(cnt);
  k_conv_x<<<4096, 256, 0, stream>>>(x, xb);
  k_conv_w1<<<2048, 256, 0, stream>>>(w1, w1t);
  k_conv_w2<<<2048, 256, 0, stream>>>(w2, w2t);
  k_gate<<<T_TOK / 8, 256, 0, stream>>>(x, gate_w, gate_b, cnt, expert_of, pos_of, wgt, token_of);
  k_prefix<<<1, 1, 0, stream>>>(cnt, base);
  k_gemm1<<<dim3(PROJ_D / 128, T_TOK / M_BLK, NEXP), 256, 0, stream>>>(xb, w1t, b1, token_of, cnt, base, hb);
  k_gemm2<<<dim3(OUT_D / 128, T_TOK / M_BLK, NEXP), 256, 0, stream>>>(hb, w2t, b2, cnt, base, ysel);
  k_combine<<<T_TOK, 256, 0, stream>>>(ysel, expert_of, pos_of, wgt, base, out);
}
